// Classifier_16913581211761
// MI455X (gfx1250) — matrix operand order repaired and checked by reading, not compiled, not run on hardware
//
#include <hip/hip_runtime.h>
#include <hip/hip_bf16.h>
#include <math.h>
#include <stdint.h>

// ---------------------------------------------------------------------------
// Problem constants (from reference)
// ---------------------------------------------------------------------------
#define NROI   256
#define PP     7
#define KBIG   43904          // 128 * 7*7*7
#define FC     1024
#define NCLS   8
#define BN_EPS 1e-3f

#define KSPLIT1 7             // 1372 K-steps = 7 * 196
#define KSTEPS1 196
#define KSPLIT2 4             // 32 K-steps = 4 * 8
#define KSTEPS2 8

typedef __bf16 v16bf __attribute__((ext_vector_type(16)));
typedef __bf16 v8bf  __attribute__((ext_vector_type(8)));
typedef float  v8f   __attribute__((ext_vector_type(8)));
typedef float  v4f   __attribute__((ext_vector_type(4)));
typedef int    v4i   __attribute__((ext_vector_type(4)));

// gfx1250 async global->LDS copy (ASYNCcnt path), with safe fallback.
#if defined(__has_builtin)
#  if __has_builtin(__builtin_amdgcn_global_load_async_to_lds_b128) && \
      __has_builtin(__builtin_amdgcn_s_wait_asynccnt)
#    define USE_ASYNC_LDS 1
#  endif
#endif
#ifndef USE_ASYNC_LDS
#  define USE_ASYNC_LDS 0
#endif

// Address-space cast helpers. The int round-trip is deliberate:
//  - AS1 (global) flat addresses are bit-identical to generic addresses.
//  - AS3 pointers are 32-bit; per CDNA5 aperture rules the LDS address is the low 32 bits of addr,
//    so the truncating inttoptr of a generic shared pointer is the LDS addr.
#define AS_GLOBAL_V4I(p) ((__attribute__((address_space(1))) v4i*)(uintptr_t)(p))
#define AS_LDS_V4I(p)    ((__attribute__((address_space(3))) v4i*)(uintptr_t)(p))

// ===========================================================================
// Kernel 1: pyramid 3D RoI align -> pooled bf16 [256][43904]
// One block per roi, 256 threads. K ordering = c*343 + z*49 + y*7 + x
// (matches w1 layout [FC][C][7][7][7] flattened over the last 4 dims).
// ===========================================================================
__global__ void __launch_bounds__(256)
roi_align_pool_kernel(const float* __restrict__ fm2,   // [128][32][32][32]
                      const float* __restrict__ fm3,   // [128][16][16][16]
                      const float* __restrict__ rois,  // [256][6]
                      __bf16* __restrict__ pooled)     // [256][43904]
{
    const int n   = blockIdx.x;
    const int tid = threadIdx.x;

    __shared__ int   s_i0[3][PP];
    __shared__ int   s_i1[3][PP];
    __shared__ float s_fr[3][PP];

    const float lo0 = rois[n*6+0], lo1 = rois[n*6+1], lo2 = rois[n*6+2];
    const float hi0 = rois[n*6+3], hi1 = rois[n*6+4], hi2 = rois[n*6+5];
    const float dd = hi0 - lo0, hh = hi1 - lo1, ww = hi2 - lo2;
    const float lvl = rintf(4.0f + __log2f(dd * hh * ww) * (1.0f / 3.0f));
    const bool  use2 = (lvl <= 2.0f);
    const float* __restrict__ fmap = use2 ? fm2 : fm3;
    const int DIM = use2 ? 32 : 16;

    if (tid < 3 * PP) {
        const int a = tid / PP;
        const int t = tid - a * PP;
        const float lo = (a == 0) ? lo0 : ((a == 1) ? lo1 : lo2);
        const float hi = (a == 0) ? hi0 : ((a == 1) ? hi1 : hi2);
        int s = (int)floorf(lo * (float)(DIM - 1));
        s = min(max(s, 0), DIM - 1);
        int e = (int)ceilf(hi * (float)(DIM - 1) + 1.0f);
        e = min(max(e, s + 1), DIM);
        const float src = (float)s + ((float)t * (1.0f / 6.0f)) * (float)(e - 1 - s);
        int i0 = min(max((int)floorf(src), 0), DIM - 1);
        int i1 = min(i0 + 1, DIM - 1);
        s_i0[a][t] = i0;
        s_i1[a][t] = i1;
        s_fr[a][t] = src - (float)i0;
    }
    __syncthreads();

    const int DD = DIM * DIM;
    for (int idx = tid; idx < KBIG; idx += 256) {
        const int c = idx / 343;
        int p = idx - c * 343;
        const int z = p / 49;  p -= z * 49;
        const int y = p / 7;
        const int x = p - y * 7;

        const int z0 = s_i0[0][z], z1 = s_i1[0][z];
        const int y0 = s_i0[1][y], y1 = s_i1[1][y];
        const int x0 = s_i0[2][x], x1 = s_i1[2][x];
        const float fz = s_fr[0][z], fy = s_fr[1][y], fx = s_fr[2][x];
        const float gz = 1.0f - fz, gy = 1.0f - fy, gx = 1.0f - fx;

        const float* __restrict__ base = fmap + (size_t)c * DD * DIM;
        const float v000 = base[(z0 * DIM + y0) * DIM + x0];
        const float v001 = base[(z0 * DIM + y0) * DIM + x1];
        const float v010 = base[(z0 * DIM + y1) * DIM + x0];
        const float v011 = base[(z0 * DIM + y1) * DIM + x1];
        const float v100 = base[(z1 * DIM + y0) * DIM + x0];
        const float v101 = base[(z1 * DIM + y0) * DIM + x1];
        const float v110 = base[(z1 * DIM + y1) * DIM + x0];
        const float v111 = base[(z1 * DIM + y1) * DIM + x1];

        const float r =
            gz * (gy * (gx * v000 + fx * v001) + fy * (gx * v010 + fx * v011)) +
            fz * (gy * (gx * v100 + fx * v101) + fy * (gx * v110 + fx * v111));

        pooled[(size_t)n * KBIG + idx] = (__bf16)r;
    }
}

// ===========================================================================
// Kernel 2/3: split-K bf16 WMMA GEMM partials.
//   part[z][n][f] = sum over K-chunk z of A[n,:] . B[f,:]
//   A: [256][K] bf16 row-major, B: [1024][K] fp32 row-major (converted to
//   bf16 inline while staging into LDS). Block: 256 threads (8 wave32s),
//   tile 128M x 64F, each wave computes 16M x 64F via 4x
//   v_wmma_f32_16x16x32_bf16 per K-step. grid = (F/64, M/128, KSPLIT).
// ===========================================================================
#define TILE_M 128
#define TILE_F 64
#define LDK    40          // 32 + 8 pad (breaks LDS bank conflicts)

__global__ void __launch_bounds__(256)
gemm_partial_kernel(const __bf16* __restrict__ A,     // [256][K] bf16
                    const float*  __restrict__ B,     // [1024][K] f32
                    float* __restrict__ part,         // [KSPLIT][256][1024] f32
                    int K, int steps)                 // steps of 32 per split
{
    __shared__ __bf16 As[TILE_M][LDK];
    __shared__ __bf16 Bs[TILE_F][LDK];

    const int tid  = threadIdx.x;
    const int wid  = tid >> 5;        // 8 wave32s
    const int lane = tid & 31;
    const int r    = lane & 15;
    const int half = lane >> 4;

    const int mBase = blockIdx.y * TILE_M;
    const int fBase = blockIdx.x * TILE_F;
    const int kBeg  = blockIdx.z * steps * 32;
    const int kEnd  = kBeg + steps * 32;

    v8f acc[4];
#pragma unroll
    for (int j = 0; j < 4; ++j)
#pragma unroll
        for (int v = 0; v < 8; ++v) acc[j][v] = 0.0f;

    // cooperative tile-load assignments
    const int aRow  = tid >> 1;          // 128 rows, 2 threads/row
    const int aKoff = (tid & 1) * 16;    // 16 bf16 per thread
    const int bRow  = tid >> 2;          // 64 rows, 4 threads/row
    const int bKoff = (tid & 3) * 8;     // 8 f32 per thread

    for (int kb = kBeg; kb < kEnd; kb += 32) {
        __syncthreads();

        // --- A tile: bf16 straight copy ---
        {
            const __bf16* asrc = A + (size_t)(mBase + aRow) * K + kb + aKoff;
#if USE_ASYNC_LDS
            // gfx1250 async DMA: global -> LDS without VGPR round-trip
            __builtin_amdgcn_global_load_async_to_lds_b128(
                AS_GLOBAL_V4I(asrc), AS_LDS_V4I(&As[aRow][aKoff]), 0, 0);
            __builtin_amdgcn_global_load_async_to_lds_b128(
                AS_GLOBAL_V4I(asrc + 8), AS_LDS_V4I(&As[aRow][aKoff + 8]), 0, 0);
#else
            const v8bf* src = (const v8bf*)asrc;
            *(v8bf*)&As[aRow][aKoff]     = src[0];
            *(v8bf*)&As[aRow][aKoff + 8] = src[1];
#endif
        }
        // --- B tile: fp32 load -> bf16 convert -> LDS ---
        {
            const float* brow = B + (size_t)(fBase + bRow) * K + kb + bKoff;
            const v4f b0 = *(const v4f*)(brow);
            const v4f b1 = *(const v4f*)(brow + 4);
            v8bf pk;
#pragma unroll
            for (int i = 0; i < 4; ++i) {
                pk[i]     = (__bf16)b0[i];
                pk[i + 4] = (__bf16)b1[i];
            }
            *(v8bf*)&Bs[bRow][bKoff] = pk;
            if (kb + 32 < kEnd)  // prefetch next B tile slice (global_prefetch_b8)
                __builtin_prefetch(brow + 32, 0, 1);
        }
#if USE_ASYNC_LDS
        __builtin_amdgcn_s_wait_asynccnt(0);
#endif
        __syncthreads();

        // --- WMMA compute ---
        // A frag (16x32, M=lane&15): K = {8h..8h+7} U {8h+16..8h+23}
        const int aR = wid * 16 + r;
        const v8bf alo = *(const v8bf*)&As[aR][half * 8];
        const v8bf ahi = *(const v8bf*)&As[aR][half * 8 + 16];
        const v16bf afrag = __builtin_shufflevector(
            alo, ahi, 0, 1, 2, 3, 4, 5, 6, 7, 8, 9, 10, 11, 12, 13, 14, 15);

#pragma unroll
        for (int j = 0; j < 4; ++j) {
            // B frag (32x16, N=lane&15): K = {8h..8h+7} U {8h+16..8h+23}
            const int col = j * 16 + r;
            const v8bf blo = *(const v8bf*)&Bs[col][half * 8];
            const v8bf bhi = *(const v8bf*)&Bs[col][half * 8 + 16];
            const v16bf bfrag = __builtin_shufflevector(
                blo, bhi, 0, 1, 2, 3, 4, 5, 6, 7, 8, 9, 10, 11, 12, 13, 14, 15);

            acc[j] = __builtin_amdgcn_wmma_f32_16x16x32_bf16(
                false, afrag, false, bfrag, (short)0, acc[j], false, false);
        }
    }

    // --- store fp32 partials ---
    // C/D layout: VGPR v -> row M = v + 8*half ; col N = lane&15
    float* pslice = part + (size_t)blockIdx.z * NROI * FC;
#pragma unroll
    for (int j = 0; j < 4; ++j) {
        const int f = fBase + j * 16 + r;
#pragma unroll
        for (int v = 0; v < 8; ++v) {
            const int nrow = mBase + wid * 16 + v + half * 8;
            pslice[(size_t)nrow * FC + f] = acc[j][v];
        }
    }
}

// ===========================================================================
// Reduction epilogue: sum split-K partials, add bias, BN, ReLU -> bf16.
// ===========================================================================
__global__ void __launch_bounds__(256)
reduce_bn_relu_kernel(const float* __restrict__ part,  // [nparts][256][1024]
                      int nparts,
                      const float* __restrict__ bias,
                      const float* __restrict__ gamma,
                      const float* __restrict__ beta,
                      const float* __restrict__ mean,
                      const float* __restrict__ var,
                      __bf16* __restrict__ out)        // [256][1024]
{
    const int idx = blockIdx.x * 256 + threadIdx.x;    // over 256*1024
    const int f   = idx & (FC - 1);

    float s = 0.0f;
    for (int z = 0; z < nparts; ++z)
        s += part[(size_t)z * NROI * FC + idx];

    const float sc = gamma[f] * rsqrtf(var[f] + BN_EPS);
    const float sh = beta[f] - mean[f] * sc;
    float t = (s + bias[f]) * sc + sh;
    t = fmaxf(t, 0.0f);
    out[idx] = (__bf16)t;
}

// ===========================================================================
// Kernel 4: heads. logits [256][8], probs [256][8], bbox [256][48].
// ===========================================================================
__global__ void __launch_bounds__(64)
heads_kernel(const __bf16* __restrict__ x2,   // [256][1024] bf16
             const float* __restrict__ wc, const float* __restrict__ bc,
             const float* __restrict__ wb, const float* __restrict__ bb,
             float* __restrict__ out)         // 2048 + 2048 + 12288 floats
{
    const int n = blockIdx.x;
    const int t = threadIdx.x;

    __shared__ float sx[FC];
    __shared__ float slog[NCLS];

    for (int i = t; i < FC; i += 64) sx[i] = (float)x2[(size_t)n * FC + i];
    __syncthreads();

    if (t < NCLS) {
        const float* wr = wc + (size_t)t * FC;
        float acc = bc[t];
        for (int k = 0; k < FC; ++k) acc += sx[k] * wr[k];
        out[(size_t)n * NCLS + t] = acc;      // logits
        slog[t] = acc;
    } else if (t < NCLS + 6 * NCLS) {
        const int j = t - NCLS;
        const float* wr = wb + (size_t)j * FC;
        float acc = bb[j];
        for (int k = 0; k < FC; ++k) acc += sx[k] * wr[k];
        out[2 * NROI * NCLS + (size_t)n * 48 + j] = acc;   // bbox
    }
    __syncthreads();

    if (t == 0) {
        float m = slog[0];
        for (int c = 1; c < NCLS; ++c) m = fmaxf(m, slog[c]);
        float e[NCLS], s = 0.0f;
        for (int c = 0; c < NCLS; ++c) { e[c] = __expf(slog[c] - m); s += e[c]; }
        const float inv = 1.0f / s;
        for (int c = 0; c < NCLS; ++c)
            out[NROI * NCLS + (size_t)n * NCLS + c] = e[c] * inv;   // probs
    }
}

// ===========================================================================
// Launcher
// ===========================================================================
extern "C" void kernel_launch(void* const* d_in, const int* in_sizes, int n_in,
                              void* d_out, int out_size, void* d_ws, size_t ws_size,
                              hipStream_t stream)
{
    const float* fm2 = (const float*)d_in[0];
    const float* fm3 = (const float*)d_in[1];
    const float* rois = (const float*)d_in[2];
    const float* w1  = (const float*)d_in[3];
    const float* b1  = (const float*)d_in[4];
    const float* g1  = (const float*)d_in[5];
    const float* be1 = (const float*)d_in[6];
    const float* m1  = (const float*)d_in[7];
    const float* v1  = (const float*)d_in[8];
    const float* w2  = (const float*)d_in[9];
    const float* b2  = (const float*)d_in[10];
    const float* g2  = (const float*)d_in[11];
    const float* be2 = (const float*)d_in[12];
    const float* m2  = (const float*)d_in[13];
    const float* v2  = (const float*)d_in[14];
    const float* wc  = (const float*)d_in[15];
    const float* bc  = (const float*)d_in[16];
    const float* wb  = (const float*)d_in[17];
    const float* bb  = (const float*)d_in[18];

    // workspace layout: pooled bf16 22.5MB | x1 bf16 0.5MB | x2 bf16 0.5MB |
    //                   fp32 split-K partials 7MB (reused by both GEMMs)
    __bf16* pooled = (__bf16*)d_ws;
    __bf16* x1 = pooled + (size_t)NROI * KBIG;
    __bf16* x2 = x1 + (size_t)NROI * FC;
    float*  parts = (float*)(x2 + (size_t)NROI * FC);

    roi_align_pool_kernel<<<NROI, 256, 0, stream>>>(fm2, fm3, rois, pooled);

    // GEMM1: [256 x 43904] @ [43904 x 1024], split-K=7 -> 224 WGs
    gemm_partial_kernel<<<dim3(FC / TILE_F, NROI / TILE_M, KSPLIT1), 256, 0, stream>>>(
        pooled, w1, parts, KBIG, KSTEPS1);
    reduce_bn_relu_kernel<<<(NROI * FC) / 256, 256, 0, stream>>>(
        parts, KSPLIT1, b1, g1, be1, m1, v1, x1);

    // GEMM2: [256 x 1024] @ [1024 x 1024], split-K=4 -> 128 WGs
    gemm_partial_kernel<<<dim3(FC / TILE_F, NROI / TILE_M, KSPLIT2), 256, 0, stream>>>(
        x1, w2, parts, FC, KSTEPS2);
    reduce_bn_relu_kernel<<<(NROI * FC) / 256, 256, 0, stream>>>(
        parts, KSPLIT2, b2, g2, be2, m2, v2, x2);

    heads_kernel<<<NROI, 64, 0, stream>>>(x2, wc, bc, wb, bb, (float*)d_out);
}
